// StableNet_18382460027354
// MI455X (gfx1250) — matrix operand order repaired and checked by reading, not compiled, not run on hardware
//
#include <hip/hip_runtime.h>
#include <stdint.h>

#define NUMF  512
#define EMB   128
#define NSTEP 3
#define LAMBDA_P 70.0f

typedef __attribute__((ext_vector_type(16))) _Float16 v16h;
typedef __attribute__((ext_vector_type(8)))  float    v8f;
typedef __attribute__((ext_vector_type(2)))  _Float16 h2;

union HFrag { v16h v; h2 p[8]; };

// ---------------- threefry2x32 (JAX-compatible), host + device ----------------
__host__ __device__ static inline uint32_t rotl32(uint32_t x, int r){
  return (x << r) | (x >> (32 - r));
}

__host__ __device__ static inline void threefry2x32(uint32_t k0, uint32_t k1,
                                                    uint32_t x0, uint32_t x1,
                                                    uint32_t* o0, uint32_t* o1){
  uint32_t k2 = k0 ^ k1 ^ 0x1BD11BDAu;
  x0 += k0; x1 += k1;
  #define TFR(r) { x0 += x1; x1 = rotl32(x1, r); x1 ^= x0; }
  TFR(13) TFR(15) TFR(26) TFR(6)
  x0 += k1; x1 += k2 + 1u;
  TFR(17) TFR(29) TFR(16) TFR(24)
  x0 += k2; x1 += k0 + 2u;
  TFR(13) TFR(15) TFR(26) TFR(6)
  x0 += k0; x1 += k1 + 3u;
  TFR(17) TFR(29) TFR(16) TFR(24)
  x0 += k1; x1 += k2 + 4u;
  TFR(13) TFR(15) TFR(26) TFR(6)
  x0 += k2; x1 += k0 + 5u;
  #undef TFR
  *o0 = x0; *o1 = x1;
}

__device__ static inline float u01_from_bits(uint32_t b){
  uint32_t f = (b >> 9) | 0x3F800000u;       // [1,2)
  return __uint_as_float(f) - 1.0f;          // [0,1)
}

// Giles (2010) single-precision erfinv
__device__ static inline float erfinv_fast(float x){
  float w = -__logf((1.0f - x) * (1.0f + x));
  float p;
  if (w < 5.0f){
    w -= 2.5f;
    p = 2.81022636e-08f;
    p = fmaf(p, w, 3.43273939e-07f);
    p = fmaf(p, w, -3.5233877e-06f);
    p = fmaf(p, w, -4.39150654e-06f);
    p = fmaf(p, w, 0.00021858087f);
    p = fmaf(p, w, -0.00125372503f);
    p = fmaf(p, w, -0.00417768164f);
    p = fmaf(p, w, 0.246640727f);
    p = fmaf(p, w, 1.50140941f);
  } else {
    w = sqrtf(w) - 3.0f;
    p = -0.000200214257f;
    p = fmaf(p, w, 0.000100950558f);
    p = fmaf(p, w, 0.00134934322f);
    p = fmaf(p, w, -0.00367342844f);
    p = fmaf(p, w, 0.00573950773f);
    p = fmaf(p, w, -0.0076224613f);
    p = fmaf(p, w, 0.00943887047f);
    p = fmaf(p, w, 1.00167406f);
    p = fmaf(p, w, 2.83297682f);
  }
  return p * x;
}

// ---------------- RNG kernel: w_rff (normals) + b_rff (uniform * 2pi) --------
__global__ void k_rng(uint32_t kw0, uint32_t kw1, uint32_t kb0, uint32_t kb1,
                      float* __restrict__ wrff, float* __restrict__ brff){
  int i = blockIdx.x * blockDim.x + threadIdx.x;
  if (i < NUMF){
    uint32_t o0, o1, bits;
    if (i < NUMF/2){ threefry2x32(kw0,kw1,(uint32_t)i,(uint32_t)(i+NUMF/2),&o0,&o1); bits=o0; }
    else           { threefry2x32(kw0,kw1,(uint32_t)(i-NUMF/2),(uint32_t)i,&o0,&o1); bits=o1; }
    const float lo = -0.99999994f;   // nextafter(-1,0)
    float u = u01_from_bits(bits) * (1.0f - lo) + lo;
    u = fmaxf(lo, u);
    wrff[i] = 1.41421356237f * erfinv_fast(u);   // sqrt(2)*erfinv, SIGMA=1
  } else {
    int j = i - NUMF;
    if (j < EMB * NUMF){
      const int H = (EMB * NUMF) / 2;
      uint32_t o0, o1, bits;
      if (j < H){ threefry2x32(kb0,kb1,(uint32_t)j,(uint32_t)(j+H),&o0,&o1); bits=o0; }
      else      { threefry2x32(kb0,kb1,(uint32_t)(j-H),(uint32_t)j,&o0,&o1); bits=o1; }
      brff[j] = 6.28318530718f * u01_from_bits(bits);
    }
  }
}

// ---------------- init / prep / update / out --------------------------------
__global__ void k_init(float* __restrict__ weight, float* __restrict__ buf){
  int t = threadIdx.x;
  weight[t] = 1.0f;
  buf[t] = 0.0f;
}

// softmax over [weight; ones(512)]: s_scaled = 1024*s_n, scal[0]=S_pre; zero grads
__global__ void k_prep(const float* __restrict__ weight,
                       float* __restrict__ sscaled,
                       float* __restrict__ scal,
                       float* __restrict__ gq,
                       float* __restrict__ gqpre){
  __shared__ float red[512];
  __shared__ float sh[2];
  const int tid = threadIdx.x;
  float wv = weight[tid];
  red[tid] = wv; __syncthreads();
  for (int s=256;s>0;s>>=1){ if (tid<s) red[tid]=fmaxf(red[tid],red[tid+s]); __syncthreads(); }
  if (tid==0) sh[0] = fmaxf(red[0], 1.0f);
  __syncthreads();
  const float mx = sh[0];
  red[tid] = __expf(wv - mx); __syncthreads();
  for (int s=256;s>0;s>>=1){ if (tid<s) red[tid]+=red[tid+s]; __syncthreads(); }
  if (tid==0) sh[1] = red[0] + 512.0f * __expf(1.0f - mx);
  __syncthreads();
  const float Z = sh[1];
  sscaled[tid] = 1024.0f * __expf(wv - mx) / Z;
  gq[tid] = 0.0f;
  if (tid==0){ scal[0] = 512.0f * __expf(1.0f - mx) / Z; *gqpre = 0.0f; }
}

__global__ void k_update(float* __restrict__ weight,
                         float* __restrict__ buf,
                         const float* __restrict__ sscaled,
                         const float* __restrict__ scal,
                         const float* __restrict__ gq,
                         const float* __restrict__ gqpre){
  __shared__ float red[512];
  __shared__ float sh[4];
  const int tid = threadIdx.x;
  const float s_n = sscaled[tid] * (1.0f/1024.0f);
  const float q_n = gq[tid];
  red[tid] = s_n * q_n; __syncthreads();
  for (int s=256;s>0;s>>=1){ if (tid<s) red[tid]+=red[tid+s]; __syncthreads(); }
  if (tid==0) sh[0] = red[0] + scal[0] * (*gqpre);
  __syncthreads();
  const float dot = sh[0];
  const float gb = s_n * (q_n - dot) * (1.0f / LAMBDA_P);
  // lossp = sum(softmax(weight)^2) gradient
  float wv = weight[tid];
  red[tid] = wv; __syncthreads();
  for (int s=256;s>0;s>>=1){ if (tid<s) red[tid]=fmaxf(red[tid],red[tid+s]); __syncthreads(); }
  if (tid==0) sh[1] = red[0]; __syncthreads();
  const float ex = __expf(wv - sh[1]);
  red[tid] = ex; __syncthreads();
  for (int s=256;s>0;s>>=1){ if (tid<s) red[tid]+=red[tid+s]; __syncthreads(); }
  if (tid==0) sh[2] = red[0]; __syncthreads();
  const float p = ex / sh[2];
  red[tid] = p*p; __syncthreads();
  for (int s=256;s>0;s>>=1){ if (tid<s) red[tid]+=red[tid+s]; __syncthreads(); }
  if (tid==0) sh[3] = red[0]; __syncthreads();
  const float gp = 2.0f*p*p - 2.0f*p*sh[3];
  const float g  = gb + gp;
  const float nb = 0.9f * buf[tid] + g;    // torch SGD momentum, lr=1
  buf[tid] = nb;
  weight[tid] = wv - nb;
}

__global__ void k_out(const float* __restrict__ weight, float* __restrict__ out){
  __shared__ float red[512];
  __shared__ float sh[2];
  const int tid = threadIdx.x;
  float wv = weight[tid];
  red[tid] = wv; __syncthreads();
  for (int s=256;s>0;s>>=1){ if (tid<s) red[tid]=fmaxf(red[tid],red[tid+s]); __syncthreads(); }
  if (tid==0) sh[0] = red[0]; __syncthreads();
  const float ex = __expf(wv - sh[0]);
  red[tid] = ex; __syncthreads();
  for (int s=256;s>0;s>>=1){ if (tid<s) red[tid]+=red[tid+s]; __syncthreads(); }
  if (tid==0) sh[1] = red[0]; __syncthreads();
  out[tid] = ex / sh[1];
}

// ---------------- main fused kernel: one workgroup per RFF slice f ----------
// LDS: XA[a=128][n=512] fp16, XN[n=512][a=128] fp16, G[128][128] fp16 (+small)
static constexpr size_t K_MAIN_LDS =
    (size_t)(128*512 + 512*128 + 128*128) * sizeof(_Float16)  // XA, XN, G
  + (size_t)(128 + 128 + 128 + 512 + 512) * sizeof(float)     // bcol,y,e,qac,ssc
  + (size_t)256 * sizeof(h2)                                  // s2
  + (size_t)256 * sizeof(float)                               // red
  + (size_t)16  * sizeof(float);                              // scalars

__global__ __launch_bounds__(256)
void k_main(const float* __restrict__ cfe,      // (512,128)
            const float* __restrict__ wrff,     // (512,)
            const float* __restrict__ brff,     // (128,512)
            const float* __restrict__ sscaled,  // 1024*s_n, n<512
            const float* __restrict__ scal,     // [0] = S_pre
            float* __restrict__ gq,             // (512,) accum dloss/ds_n
            float* __restrict__ gqpre){         // accum dloss/ds_pre
  extern __shared__ char smem[];
  _Float16* XA = (_Float16*)smem;               // [a][n]
  _Float16* XN = XA + 128*512;                  // [n][a]
  _Float16* Gm = XN + 512*128;                  // [a][b], fp16 * 256
  float* bcol = (float*)(Gm + 128*128);         // 128
  float* yv   = bcol + 128;                     // 128
  float* ev   = yv + 128;                       // 128
  float* qac  = ev + 128;                       // 512
  float* ssc  = qac + 512;                      // 512
  h2*    s2   = (h2*)(ssc + 512);               // 256 packed (1024*s) pairs
  float* red  = (float*)(s2 + 256);             // 256
  float* shsc = red + 256;

  const int tid  = threadIdx.x;
  const int f    = blockIdx.x;
  const int w    = tid >> 5;          // wave id 0..7 (wave32)
  const int lane = tid & 31;
  const int l15  = lane & 15;
  const int hi8  = (lane >= 16) ? 8 : 0;
  const int khB  = (lane >= 16) ? 16 : 0;

  int koffA[8];
  #pragma unroll
  for (int v = 0; v < 8; ++v)
    koffA[v] = ((v < 4) ? (2*v) : (16 + 2*(v-4))) + hi8;

  // ---- phase 0: small loads
  for (int n = tid; n < 512; n += 256){ ssc[n] = sscaled[n]; qac[n] = 0.0f; }
  if (tid < 128) bcol[tid] = brff[tid * NUMF + f];
  if (tid == 0)  shsc[0] = scal[0];
  __syncthreads();

  { h2 t; t.x = (_Float16)ssc[2*tid]; t.y = (_Float16)ssc[2*tid+1]; s2[tid] = t; }

  // pre-row feature y[a] (all-zero pre_features => mid = b column)
  if (tid == 0){
    float mn = 1e30f, mx = -1e30f;
    for (int a = 0; a < 128; ++a){ float m = bcol[a]; mn = fminf(mn,m); mx = fmaxf(mx,m); }
    float inv = 1.57079632679f / (mx - mn);
    for (int a = 0; a < 128; ++a){
      float m = (bcol[a] - mn) * inv;
      yv[a] = 0.0625f * (__cosf(m) + __sinf(m));   // sqrt(2/512)=1/16
    }
  }

  // ---- phase A: features x[n][a] for n<512 into XA and XN (fp16)
  const float wf = wrff[f];
  for (int rr = 0; rr < 2; ++rr){
    const int n = tid + 256*rr;
    const float* crow = cfe + n * EMB;
    __builtin_prefetch(crow, 0, 3);
    float mn = 1e30f, mx = -1e30f;
    for (int a = 0; a < 128; ++a){
      float m = fmaf(crow[a], wf, bcol[a]);
      mn = fminf(mn, m); mx = fmaxf(mx, m);
    }
    float inv = 1.57079632679f / (mx - mn);
    for (int a = 0; a < 128; ++a){
      float m = (fmaf(crow[a], wf, bcol[a]) - mn) * inv;
      _Float16 h = (_Float16)(0.0625f * (__cosf(m) + __sinf(m)));
      XA[a*512 + n] = h;
      XN[n*128 + a] = h;
    }
  }
  __syncthreads();

  // ---- weighted mean e[a] = sum_n s_n x[n,a] + S_pre*y[a]
  const float Spre = shsc[0];
  if (tid < 128){
    float s = 0.0f;
    for (int n = 0; n < 512; ++n) s += ssc[n] * (float)XA[tid*512 + n];
    ev[tid] = s * (1.0f/1024.0f) + Spre * yv[tid];
  }
  __syncthreads();

  // ---- phase B: cov = (S*X)^T X via WMMA; wave w owns row-tile a=[16w,16w+16)
  const int arow = 16*w + l15;
  auto loadA = [&](HFrag& A, int nb){
    #pragma unroll
    for (int v = 0; v < 8; ++v){
      const int n = nb + koffA[v];
      h2 x = *(const h2*)&XA[arow*512 + n];
      A.p[v] = x * s2[n >> 1];                   // fold 1024*s_n into A
    }
  };
  auto loadB = [&](HFrag& B, int nb, int t){
    const int bc = 16*t + l15;
    #pragma unroll
    for (int v = 0; v < 8; ++v)
      B.p[v] = *(const h2*)&XA[bc*512 + nb + koffA[v]];
  };

  v8f acc[8] = {};
  {
    HFrag Af[2], Bf[2];
    loadA(Af[0], 0);
    loadB(Bf[0], 0, 0);
    #pragma unroll 2
    for (int c = 0; c < 16; ++c){
      const int nb = 32*c;
      if (c < 15) loadA(Af[(c+1)&1], nb + 32);   // prefetch next K-chunk A
      #pragma unroll
      for (int t = 0; t < 8; ++t){
        if (t < 7)       loadB(Bf[(t+1)&1], nb, t+1);       // prefetch next tile B
        else if (c < 15) loadB(Bf[0],       nb + 32, 0);    // first B of next chunk
        acc[t] = __builtin_amdgcn_wmma_f32_16x16x32_f16(
            false, Af[c&1].v, false, Bf[t&1].v, (short)0, acc[t], false, false);
      }
    }
  }
  // G = 2*(cov + S_pre*yy^T - ee^T) with zero diagonal, fp16 * 256
  #pragma unroll
  for (int t = 0; t < 8; ++t){
    const int b = 16*t + l15;
    const float yb = yv[b], eb = ev[b];
    #pragma unroll
    for (int v = 0; v < 8; ++v){
      const int a = 16*w + v + hi8;
      float m = acc[t][v] * (1.0f/1024.0f) + Spre * yv[a] * yb - ev[a] * eb;
      float g = (a == b) ? 0.0f : (2.0f * m * 256.0f);
      Gm[a*128 + b] = (_Float16)g;
    }
  }
  __syncthreads();

  // ---- U = X - 2e (in place on XA; XA no longer needed raw)
  for (int i = tid; i < 128*256; i += 256){
    const int a = i >> 8;
    h2 x = ((h2*)XA)[i];
    _Float16 e2 = (_Float16)(2.0f * ev[a]);
    h2 d; d.x = x.x - e2; d.y = x.y - e2;
    ((h2*)XA)[i] = d;
  }
  __syncthreads();

  // ---- phase D: H = G*X (WMMA), q_n = sum_a U[a,n]*H[a,n] -> qac via ds_add
  // ta outer so the 4 G fragments (A operand) are loaded once and reused
  // across the wave's 4 n-tiles.
  {
    float qp[4] = {0.0f, 0.0f, 0.0f, 0.0f};
    for (int ta = 0; ta < 8; ++ta){
      const int am = 16*ta + l15;
      HFrag GA[4];
      #pragma unroll
      for (int c = 0; c < 4; ++c){
        #pragma unroll
        for (int v = 0; v < 8; ++v)
          GA[c].p[v] = *(const h2*)&Gm[am*128 + 32*c + koffA[v]];
      }
      #pragma unroll
      for (int i = 0; i < 4; ++i){
        const int ncol = 16*(w + 8*i) + l15;     // sample index
        v8f hacc = {};
        #pragma unroll
        for (int c = 0; c < 4; ++c){
          HFrag B;
          #pragma unroll
          for (int v = 0; v < 8; ++v)
            B.p[v] = *(const h2*)&XN[ncol*128 + 32*c + koffA[v]];
          hacc = __builtin_amdgcn_wmma_f32_16x16x32_f16(
              false, GA[c].v, false, B.v, (short)0, hacc, false, false);
        }
        #pragma unroll
        for (int v = 0; v < 8; ++v){
          const int a2 = 16*ta + v + hi8;
          qp[i] += (float)XA[a2*512 + ncol] * hacc[v];   // U * H
        }
      }
    }
    #pragma unroll
    for (int i = 0; i < 4; ++i)
      atomicAdd(&qac[16*(w + 8*i) + l15], qp[i] * (1.0f/256.0f));  // undo G scale
  }
  __syncthreads();

  // ---- q_pre = (y-2e)^T G y  (per f), reduce across block
  float contrib = 0.0f;
  if (tid < 128){
    float hy = 0.0f;
    for (int b = 0; b < 128; ++b) hy += (float)Gm[tid*128 + b] * yv[b];
    contrib = (yv[tid] - 2.0f*ev[tid]) * hy * (1.0f/256.0f);
  }
  red[tid] = contrib;
  __syncthreads();
  for (int s = 128; s > 0; s >>= 1){
    if (tid < s) red[tid] += red[tid + s];
    __syncthreads();
  }
  if (tid == 0) atomicAdd(gqpre, red[0]);
  for (int n = tid; n < 512; n += 256) atomicAdd(&gq[n], qac[n]);
}

// ---------------- host side --------------------------------------------------
extern "C" void kernel_launch(void* const* d_in, const int* in_sizes, int n_in,
                              void* d_out, int out_size, void* d_ws, size_t ws_size,
                              hipStream_t stream) {
  (void)in_sizes; (void)n_in; (void)out_size; (void)ws_size;
  const float* cfe = (const float*)d_in[0];   // cfeatures (512,128)
  float* out = (float*)d_out;                 // (512,)
  float* ws  = (float*)d_ws;
  float* weight  = ws;            // 512
  float* buf     = ws + 512;      // 512
  float* sscaled = ws + 1024;     // 512
  float* gq      = ws + 1536;     // 512
  float* scal    = ws + 2048;     // [0]=S_pre, [1]=gq_pre
  float* wrff    = ws + 2064;     // 512
  float* brff    = ws + 2576;     // 65536

  // JAX key schedule on host: key(1) = (0,1); split into 3 step keys.
  uint32_t y0[3], y1[3];
  for (uint32_t i = 0; i < 3; ++i)
    threefry2x32(0u, 1u, i, i + 3u, &y0[i], &y1[i]);
  uint32_t out6[6] = { y0[0], y0[1], y0[2], y1[0], y1[1], y1[2] };

  k_init<<<1, 512, 0, stream>>>(weight, buf);

  for (int t = 0; t < NSTEP; ++t){
    uint32_t sk0 = out6[2*t], sk1 = out6[2*t+1];
    // split(step_key, 2): counts iota(4) -> pairs (0,2),(1,3)
    uint32_t a0, b0, a1, b1;
    threefry2x32(sk0, sk1, 0u, 2u, &a0, &b0);
    threefry2x32(sk0, sk1, 1u, 3u, &a1, &b1);
    uint32_t kw0 = a0, kw1 = a1;   // first row of reshape(2,2)
    uint32_t kb0 = b0, kb1 = b1;   // second row

    k_rng<<<(NUMF + EMB*NUMF) / 256, 256, 0, stream>>>(kw0, kw1, kb0, kb1, wrff, brff);
    k_prep<<<1, 512, 0, stream>>>(weight, sscaled, scal, gq, scal + 1);
    k_main<<<NUMF, 256, K_MAIN_LDS, stream>>>(cfe, wrff, brff, sscaled, scal, gq, scal + 1);
    k_update<<<1, 512, 0, stream>>>(weight, buf, sscaled, scal, gq, scal + 1);
  }
  k_out<<<1, 512, 0, stream>>>(weight, out);
}
